// MotionInjectionProcessor_65429531787380
// MI455X (gfx1250) — matrix operand order repaired and checked by reading, not compiled, not run on hardware
//
#include <hip/hip_runtime.h>
#include <hip/hip_bf16.h>

typedef __bf16 bf16_t;
typedef __attribute__((ext_vector_type(16))) __bf16 v16bf;
typedef __attribute__((ext_vector_type(8)))  __bf16 v8bf;
typedef __attribute__((ext_vector_type(8)))  float  v8f;

#define B_     2
#define SEQ_   2048
#define DTOK_  1536
#define H_     24
#define DH_    64
#define TP_    4
#define NS_    256
#define NMH_   8
#define DFLOW_ 64

#if defined(__gfx1250__) && __has_builtin(__builtin_amdgcn_global_load_async_to_lds_b128)
#define HAVE_ASYNC 1
#else
#define HAVE_ASYNC 0
#endif

// The async builtin takes (int4 AS1* src, int4 AS3* dst, imm offset, imm cpol)
typedef int v4i_t __attribute__((vector_size(16)));
typedef __attribute__((address_space(1))) v4i_t* g4_t;
typedef __attribute__((address_space(3))) v4i_t* l4_t;

__device__ __forceinline__ bf16_t f2bf(float f) {
  unsigned u = __builtin_bit_cast(unsigned, f);
  u += 0x7FFFu + ((u >> 16) & 1u);           // round-to-nearest-even
  unsigned short h = (unsigned short)(u >> 16);
  return __builtin_bit_cast(bf16_t, h);
}

__device__ __forceinline__ v16bf cat8(v8bf lo, v8bf hi) {
  return __builtin_shufflevector(lo, hi, 0,1,2,3,4,5,6,7,8,9,10,11,12,13,14,15);
}

// 16-byte global -> LDS copy; async (ASYNCcnt) when the toolchain exposes it.
// AS1 pointer value == flat address; AS3 pointer value == low 32 bits of the
// flat LDS address (the LDS offset, per the flat-aperture mapping).
__device__ __forceinline__ void cp16_async(void* lds, const void* g) {
#if HAVE_ASYNC
  __builtin_amdgcn_global_load_async_to_lds_b128(
      (g4_t)(unsigned long long)g,
      (l4_t)(unsigned)(unsigned long long)lds,
      0, 0);
#else
  *(v8bf*)lds = *(const v8bf*)g;
#endif
}

__device__ __forceinline__ void wait_async() {
#if HAVE_ASYNC
#if __has_builtin(__builtin_amdgcn_s_wait_asynccnt)
  __builtin_amdgcn_s_wait_asynccnt(0);
#else
  asm volatile("s_wait_asynccnt 0x0" ::: "memory");
#endif
#endif
}

// ---------------------------------------------------------------------------
// dK/dV: dX[b,h,row,d] = gate[b,row] * sum_f M[b,row,f] * F[h,f,d]
// ---------------------------------------------------------------------------
__global__ __launch_bounds__(64)
void dkv_kernel(const float* __restrict__ M, const float* __restrict__ Wreg,
                const float* __restrict__ Fk, const float* __restrict__ Fv,
                float* __restrict__ dK, float* __restrict__ dV)
{
  __shared__ float Mrow[DFLOW_];
  const int blk = blockIdx.x;
  const int b   = blk / (NMH_ * TP_ * NS_);
  const int rem = blk % (NMH_ * TP_ * NS_);
  const int h   = rem / (TP_ * NS_);
  const int row = rem % (TP_ * NS_);
  const int d   = threadIdx.x;

  Mrow[d] = M[((size_t)b * TP_ * NS_ + row) * DFLOW_ + d];
  __syncthreads();
  const float gate = Wreg[b * TP_ * NS_ + row];

  float ak = 0.f, av = 0.f;
  #pragma unroll 8
  for (int f = 0; f < DFLOW_; ++f) {
    const float m = Mrow[f];
    ak += m * Fk[((size_t)h * DFLOW_ + f) * DH_ + d];
    av += m * Fv[((size_t)h * DFLOW_ + f) * DH_ + d];
  }
  const size_t off = ((size_t)(b * NMH_ + h) * (TP_ * NS_) + row) * DH_ + d;
  dK[off] = ak * gate;
  dV[off] = av * gate;
}

// ---------------------------------------------------------------------------
// Projection GEMM: C[4096x1536] = A[4096x1536] * W[1536x1536] (+epilogue)
// MODE 0: Q -> bf16 [B,H,SEQ,DH]
// MODE 1: K -> bf16 [B,H,SEQ,DH], += dK for h < NMH
// MODE 2: V -> bf16 [B,H,DH,SEQ] (transposed), += dV for h < NMH
// MODE 3: out = A*Wo + bo -> fp32 [B,SEQ,DTOK]
// ---------------------------------------------------------------------------
#define BM 128
#define BN 128
#define BK 32
#define LDA_ 40   // padded halfword stride (bank-conflict free, 16B aligned rows)

template<int MODE>
__global__ __launch_bounds__(256)
void proj_kernel(const float* __restrict__ A, const float* __restrict__ W,
                 const float* __restrict__ dAdd, const float* __restrict__ bias,
                 bf16_t* __restrict__ outH, float* __restrict__ outF)
{
  __shared__ bf16_t As[BM * LDA_];
  __shared__ bf16_t Bs[BN * LDA_];
  const int tid  = threadIdx.x;
  const int lane = tid & 31;
  const int wid  = tid >> 5;
  const int wm   = wid >> 2;       // 0..1
  const int wn   = wid & 3;        // 0..3
  const int half = lane >> 4;
  const int l16  = lane & 15;
  const int row0 = blockIdx.x * BM;
  const int col0 = blockIdx.y * BN;
  const int KD = DTOK_, NC = DTOK_;

  v8f acc[4][2];
  #pragma unroll
  for (int i = 0; i < 4; ++i)
    #pragma unroll
    for (int j = 0; j < 2; ++j)
      acc[i][j] = (v8f){0.f,0.f,0.f,0.f,0.f,0.f,0.f,0.f};

  for (int k0 = 0; k0 < KD; k0 += BK) {
    // Stage A tile 128x32 (fp32 -> bf16), row-major [m][k]
    #pragma unroll
    for (int i = 0; i < 4; ++i) {
      const int lin = tid + i * 256;
      const int r   = lin >> 3;
      const int c4  = (lin & 7) << 2;
      const float4 v = *(const float4*)&A[(size_t)(row0 + r) * KD + k0 + c4];
      bf16_t* dst = &As[r * LDA_ + c4];
      dst[0] = f2bf(v.x); dst[1] = f2bf(v.y); dst[2] = f2bf(v.z); dst[3] = f2bf(v.w);
    }
    // Stage B tile: W[k][n] stored transposed Bs[n][k]
    #pragma unroll
    for (int i = 0; i < 4; ++i) {
      const int lin = tid + i * 256;
      const int kr  = lin >> 5;
      const int n4  = (lin & 31) << 2;
      const float4 v = *(const float4*)&W[(size_t)(k0 + kr) * NC + col0 + n4];
      Bs[(n4 + 0) * LDA_ + kr] = f2bf(v.x);
      Bs[(n4 + 1) * LDA_ + kr] = f2bf(v.y);
      Bs[(n4 + 2) * LDA_ + kr] = f2bf(v.z);
      Bs[(n4 + 3) * LDA_ + kr] = f2bf(v.w);
    }
    if (k0 + BK < KD) {   // hint next W tile toward the caches
      __builtin_prefetch(&W[(size_t)(k0 + BK + (tid & 31)) * NC + col0 + (tid >> 5) * 16], 0, 1);
    }
    __syncthreads();

    v16bf af[4], bfr[2];
    #pragma unroll
    for (int im = 0; im < 4; ++im) {
      const bf16_t* p = &As[(wm * 64 + im * 16 + l16) * LDA_ + 8 * half];
      af[im] = cat8(*(const v8bf*)p, *(const v8bf*)(p + 16));
    }
    #pragma unroll
    for (int in_ = 0; in_ < 2; ++in_) {
      const bf16_t* p = &Bs[(wn * 32 + in_ * 16 + l16) * LDA_ + 8 * half];
      bfr[in_] = cat8(*(const v8bf*)p, *(const v8bf*)(p + 16));
    }
    #pragma unroll
    for (int im = 0; im < 4; ++im)
      #pragma unroll
      for (int in_ = 0; in_ < 2; ++in_)
        acc[im][in_] = __builtin_amdgcn_wmma_f32_16x16x32_bf16(
            false, af[im], false, bfr[in_], (short)0, acc[im][in_], false, false);
    __syncthreads();
  }

  // Epilogue: C layout -> lane l holds n=l%16, vgpr v holds m=v+8*(l/16)
  #pragma unroll
  for (int im = 0; im < 4; ++im) {
    #pragma unroll
    for (int in_ = 0; in_ < 2; ++in_) {
      #pragma unroll
      for (int v = 0; v < 8; ++v) {
        const int r = row0 + wm * 64 + im * 16 + v + 8 * half;
        const int c = col0 + wn * 32 + in_ * 16 + l16;
        float val = acc[im][in_][v];
        if constexpr (MODE == 3) {
          outF[(size_t)r * NC + c] = val + bias[c];
        } else {
          const int b = r >> 11, s = r & (SEQ_ - 1);
          const int h = c >> 6,  d = c & 63;
          if constexpr (MODE != 0) {
            if (h < NMH_)
              val += dAdd[(((b * NMH_ + h) << 10) + (s & 1023)) * DH_ + d];
          }
          if constexpr (MODE == 2)
            outH[((size_t)(b * H_ + h) * DH_ + d) * SEQ_ + s] = f2bf(val);   // V^T
          else
            outH[((size_t)(b * H_ + h) * SEQ_ + s) * DH_ + d] = f2bf(val);   // Q,K
        }
      }
    }
  }
}

// ---------------------------------------------------------------------------
// Flash attention: block = (b, h, 128-query tile), 8 waves x 16 query rows.
// K / V^T 64-key tiles are cooperatively streamed into double-buffered LDS
// with GLOBAL_LOAD_ASYNC_TO_LDS_B128 (ASYNCcnt), overlapping the WMMA sweep
// of the current tile. One barrier per key block.
// ---------------------------------------------------------------------------
#define KST 72   // padded LDS stride (bank-conflict free, rows 16B aligned)

__global__ __launch_bounds__(256)
void attn_kernel(const bf16_t* __restrict__ Qb, const bf16_t* __restrict__ Kb,
                 const bf16_t* __restrict__ VTb, float* __restrict__ AO)
{
  __shared__ bf16_t Ks[2][64 * KST];
  __shared__ bf16_t Vs[2][64 * KST];
  __shared__ bf16_t Pld[8][16 * KST];
  const int tid = threadIdx.x, lane = tid & 31, wid = tid >> 5;
  const int half = lane >> 4, l16 = lane & 15;
  const int qt = blockIdx.x, h = blockIdx.y, b = blockIdx.z;
  const size_t base = (size_t)(b * H_ + h) * SEQ_ * DH_;
  const bf16_t* qp = Qb + base;
  const bf16_t* kp = Kb + base;
  const bf16_t* vp = VTb + base;    // [DH][SEQ]
  const int qrow0 = qt * 128 + wid * 16;
  bf16_t* P = Pld[wid];

  // cooperative 64-key tile stage: 64 rows x 128B, 16B chunks, 2+2 per thread
  auto stage = [&](int key0, int buf) {
    #pragma unroll
    for (int i = 0; i < 2; ++i) {
      const int lin = tid + i * 256;       // 0..511
      const int row = lin >> 3;
      const int c16 = (lin & 7) << 3;      // element offset (8 bf16 = 16B)
      cp16_async(&Ks[buf][row * KST + c16], kp + (size_t)(key0 + row) * DH_ + c16);
      cp16_async(&Vs[buf][row * KST + c16], vp + (size_t)row * SEQ_ + key0 + c16);
    }
  };

  // Q A-fragments (held for the whole key sweep)
  v16bf aq[2];
  #pragma unroll
  for (int c = 0; c < 2; ++c) {
    const bf16_t* p = qp + (size_t)(qrow0 + l16) * DH_ + c * 32 + 8 * half;
    aq[c] = cat8(*(const v8bf*)p, *(const v8bf*)(p + 16));
  }

  float rmax[8], rsum[8];
  v8f o[4];
  #pragma unroll
  for (int v = 0; v < 8; ++v) { rmax[v] = -3.0e38f; rsum[v] = 0.f; }
  #pragma unroll
  for (int t = 0; t < 4; ++t) o[t] = (v8f){0.f,0.f,0.f,0.f,0.f,0.f,0.f,0.f};

  stage(0, 0);
  for (int kb = 0; kb < SEQ_ / 64; ++kb) {
    const int cur = kb & 1;
    wait_async();                 // this wave's chunks of tile kb resident
    __syncthreads();              // all waves' chunks resident; prev buf free
    if (kb + 1 < SEQ_ / 64) stage((kb + 1) * 64, cur ^ 1);

    // S = Q K^T  (16 x 64 per wave) from LDS
    v8f s4[4];
    #pragma unroll
    for (int j = 0; j < 4; ++j) s4[j] = (v8f){0.f,0.f,0.f,0.f,0.f,0.f,0.f,0.f};
    #pragma unroll
    for (int j = 0; j < 4; ++j) {
      const bf16_t* kbp = &Ks[cur][(j * 16 + l16) * KST + 8 * half];
      #pragma unroll
      for (int c = 0; c < 2; ++c) {
        v16bf bk = cat8(*(const v8bf*)(kbp + c * 32), *(const v8bf*)(kbp + c * 32 + 16));
        s4[j] = __builtin_amdgcn_wmma_f32_16x16x32_bf16(
            false, aq[c], false, bk, (short)0, s4[j], false, false);
      }
    }
    // Online softmax: rows live as (v, half); reduce across lanes 0..15 per half
    #pragma unroll
    for (int v = 0; v < 8; ++v) {
      float mx = fmaxf(fmaxf(s4[0][v], s4[1][v]), fmaxf(s4[2][v], s4[3][v]));
      mx = fmaxf(mx, __shfl_xor(mx, 1, 32));
      mx = fmaxf(mx, __shfl_xor(mx, 2, 32));
      mx = fmaxf(mx, __shfl_xor(mx, 4, 32));
      mx = fmaxf(mx, __shfl_xor(mx, 8, 32));
      mx *= 0.125f;                               // 1/sqrt(DH)
      const float nm = fmaxf(rmax[v], mx);
      const float corr = __expf(rmax[v] - nm);
      rmax[v] = nm;
      float bs = 0.f;
      #pragma unroll
      for (int j = 0; j < 4; ++j) {
        const float p = __expf(s4[j][v] * 0.125f - nm);
        bs += p;
        P[(v + 8 * half) * KST + j * 16 + l16] = f2bf(p);
      }
      bs += __shfl_xor(bs, 1, 32);
      bs += __shfl_xor(bs, 2, 32);
      bs += __shfl_xor(bs, 4, 32);
      bs += __shfl_xor(bs, 8, 32);
      rsum[v] = rsum[v] * corr + bs;
      #pragma unroll
      for (int t = 0; t < 4; ++t) o[t][v] *= corr;
    }
    // P A-fragments from LDS (per-wave region; DS ops are in-order per wave)
    v16bf ap[2];
    #pragma unroll
    for (int c = 0; c < 2; ++c) {
      const bf16_t* p = &P[l16 * KST + c * 32 + 8 * half];
      ap[c] = cat8(*(const v8bf*)p, *(const v8bf*)(p + 16));
    }
    // O += P V from LDS V^T tile
    #pragma unroll
    for (int t = 0; t < 4; ++t) {
      const bf16_t* vb = &Vs[cur][(t * 16 + l16) * KST + 8 * half];
      #pragma unroll
      for (int c = 0; c < 2; ++c) {
        v16bf bv = cat8(*(const v8bf*)(vb + c * 32), *(const v8bf*)(vb + c * 32 + 16));
        o[t] = __builtin_amdgcn_wmma_f32_16x16x32_bf16(
            false, ap[c], false, bv, (short)0, o[t], false, false);
      }
    }
  }
  // Normalize and write [B,SEQ,H,DH] (== rows of the final GEMM input)
  #pragma unroll
  for (int v = 0; v < 8; ++v) {
    const float inv = 1.f / rsum[v];
    const int srow = qrow0 + v + 8 * half;
    #pragma unroll
    for (int t = 0; t < 4; ++t)
      AO[(size_t)(b * SEQ_ + srow) * DTOK_ + h * DH_ + t * 16 + l16] = o[t][v] * inv;
  }
}

// ---------------------------------------------------------------------------
extern "C" void kernel_launch(void* const* d_in, const int* in_sizes, int n_in,
                              void* d_out, int out_size, void* d_ws, size_t ws_size,
                              hipStream_t stream) {
  const float* hs = (const float*)d_in[0];
  const float* Wq = (const float*)d_in[1];
  const float* Wk = (const float*)d_in[2];
  const float* Wv = (const float*)d_in[3];
  const float* Wo = (const float*)d_in[4];
  const float* bo = (const float*)d_in[5];
  const float* M  = (const float*)d_in[6];
  const float* Wr = (const float*)d_in[7];
  const float* Fk = (const float*)d_in[8];
  const float* Fv = (const float*)d_in[9];
  float* out = (float*)d_out;

  char* w = (char*)d_ws;
  const size_t dkBytes  = (size_t)B_ * NMH_ * TP_ * NS_ * DH_ * sizeof(float); // 4 MiB
  const size_t qkvBytes = (size_t)B_ * H_ * SEQ_ * DH_ * sizeof(bf16_t);       // 12 MiB
  float*  dK = (float*)(w);
  float*  dV = (float*)(w + dkBytes);
  bf16_t* Qb = (bf16_t*)(w + 2 * dkBytes);
  bf16_t* Kb = (bf16_t*)(w + 2 * dkBytes + qkvBytes);
  bf16_t* VT = (bf16_t*)(w + 2 * dkBytes + 2 * qkvBytes);
  float*  AO = (float*)(w + 2 * dkBytes + 3 * qkvBytes);

  dkv_kernel<<<B_ * NMH_ * TP_ * NS_, 64, 0, stream>>>(M, Wr, Fk, Fv, dK, dV);

  dim3 pgrid(B_ * SEQ_ / BM, DTOK_ / BN);
  proj_kernel<0><<<pgrid, 256, 0, stream>>>(hs, Wq, nullptr, nullptr, Qb, nullptr);
  proj_kernel<1><<<pgrid, 256, 0, stream>>>(hs, Wk, dK, nullptr, Kb, nullptr);
  proj_kernel<2><<<pgrid, 256, 0, stream>>>(hs, Wv, dV, nullptr, VT, nullptr);

  attn_kernel<<<dim3(SEQ_ / 128, H_, B_), 256, 0, stream>>>(Qb, Kb, VT, AO);

  proj_kernel<3><<<pgrid, 256, 0, stream>>>(AO, Wo, nullptr, bo, nullptr, out);
}
